// RegionAttntion_11751030522444
// MI455X (gfx1250) — matrix operand order repaired and checked by reading, not compiled, not run on hardware
//
#include <hip/hip_runtime.h>
#include <hip/hip_bf16.h>
#include <stdint.h>

// ---------------- problem constants ----------------
#define B_BATCH   8
#define L_TOK     4096
#define DIM       512
#define HEADS     8
#define HD        64
#define REGION    16
#define NTOK      256            // tokens per window
#define NWIN      128            // B * 16 windows
#define CONV_K    15
#define QKV_N     1536
#define M_TOT     (NWIN * NTOK)  // 32768 window-ordered tokens
#define SCALE     0.125f

typedef __attribute__((ext_vector_type(16))) _Float16 v16h;
typedef __attribute__((ext_vector_type(8)))  float    v8f;
typedef __attribute__((ext_vector_type(4)))  uint32_t u32x4;
typedef __attribute__((ext_vector_type(8)))  uint32_t u32x8;

// ---------------- WMMA helpers ----------------
__device__ __forceinline__ v8f wmma32(v16h a, v16h b, v8f c) {
  return __builtin_amdgcn_wmma_f32_16x16x32_f16(false, a, false, b, (short)0, c,
                                                false, false);
}

__device__ __forceinline__ v8f zero8() {
  v8f z;
#pragma unroll
  for (int i = 0; i < 8; ++i) z[i] = 0.0f;
  return z;
}

// ---------------- CDNA5 data movers ----------------
// TDM: copy `nelem` contiguous f16 (nelem <= 65535) from global to LDS.
// D# group0/group1 built in SGPRs (all inputs wave-uniform). 1-row 2D tile.
__device__ __forceinline__ void tdm_copy_f16(void* lds_dst,
                                             const _Float16* gsrc,
                                             uint32_t nelem) {
  uint32_t lds_off = (uint32_t)(uintptr_t)lds_dst;   // low 32 bits = LDS offset
  uint64_t ga = (uint64_t)(uintptr_t)gsrc;
  u32x4 g0;
  g0[0] = 1u;                                         // count=1, user mode
  g0[1] = lds_off;                                    // lds_addr
  g0[2] = (uint32_t)ga;                               // global_addr[31:0]
  g0[3] = (uint32_t)((ga >> 32) & 0x01FFFFFFu)        // global_addr[56:32]
          | 0x80000000u;                              // type=2 (image)
  u32x8 g1;
  g1[0] = 1u << 16;                                   // data_size=1 (2 bytes)
  g1[1] = (nelem & 0xFFFFu) << 16;                    // tensor_dim0[15:0]
  g1[2] = (nelem >> 16) | (1u << 16);                 // tensor_dim0[31:16] | tensor_dim1=1
  g1[3] = (nelem & 0xFFFFu) << 16;                    // tile_dim0 = nelem
  g1[4] = 1u;                                         // tile_dim1=1, tile_dim2=0
  g1[5] = nelem;                                      // tensor_dim0_stride[31:0]
  g1[6] = 0u;                                         // stride0 hi | stride1 lo
  g1[7] = 0u;                                         // stride1 hi
  asm volatile("tensor_load_to_lds %0, %1" ::"s"(g0), "s"(g1) : "memory");
}

__device__ __forceinline__ void tdm_wait0() {
  __builtin_amdgcn_s_wait_tensorcnt(0);
}

// per-lane async global->LDS 16-byte copy (GLOBAL_LOAD_ASYNC_TO_LDS_B128, GV mode)
__device__ __forceinline__ void async_g2l_b128(void* lds_dst, const void* gsrc) {
  uint32_t l = (uint32_t)(uintptr_t)lds_dst;
  uint64_t g = (uint64_t)(uintptr_t)gsrc;
  asm volatile("global_load_async_to_lds_b128 %0, %1, off" ::"v"(l), "v"(g)
               : "memory");
}

__device__ __forceinline__ void async_wait0() {
  asm volatile("s_wait_asynccnt 0" ::: "memory");
}

// ---------------- fragment loaders ----------------
// A-fragment (16x32 f16) from row-major LDS tile [rows][ld].
__device__ __forceinline__ v16h lda16(const _Float16* base, int row, int ld,
                                      int k0, int lane) {
  int r  = row + (lane & 15);
  int kh = (lane >> 4) << 3;
  const uint32_t* p0 = (const uint32_t*)(base + r * ld + k0 + kh);
  const uint32_t* p1 = (const uint32_t*)(base + r * ld + k0 + kh + 16);
  union { v16h v; uint32_t u[8]; } f;
#pragma unroll
  for (int i = 0; i < 4; ++i) { f.u[i] = p0[i]; f.u[4 + i] = p1[i]; }
  return f.v;
}

// B-fragment (32x16 f16). LDS tile stored as [col][K]; per-lane K = kh..kh+7, kh+16..kh+23.
__device__ __forceinline__ v16h ldb16(const _Float16* base, int col, int ld,
                                      int k0, int lane) {
  int c  = col + (lane & 15);
  int kh = (lane >> 4) << 3;
  const uint32_t* p0 = (const uint32_t*)(base + c * ld + k0 + kh);
  const uint32_t* p1 = (const uint32_t*)(base + c * ld + k0 + kh + 16);
  union { v16h v; uint32_t u[8]; } f;
#pragma unroll
  for (int i = 0; i < 4; ++i) { f.u[i] = p0[i]; f.u[4 + i] = p1[i]; }
  return f.v;
}

// A-fragment built from f32 LDS (softmaxed scores), f16 convert on the fly.
__device__ __forceinline__ v16h lda32(const float* S, int row, int ld, int k0,
                                      int lane) {
  int r  = row + (lane & 15);
  int kh = (lane >> 4) << 3;
  const float* p0 = S + r * ld + k0 + kh;
  const float* p1 = p0 + 16;
  v16h f;
#pragma unroll
  for (int i = 0; i < 8; ++i) {
    f[i]     = (_Float16)p0[i];
    f[8 + i] = (_Float16)p1[i];
  }
  return f;
}

// C/D layout: VGPR r, lanes 0-15 -> (M=row+r, N=lane); lanes 16-31 -> (M=row+8+r)
__device__ __forceinline__ void stacc(float* dst, int row, int ld, int col,
                                      int lane, v8f c) {
  int n  = col + (lane & 15);
  int r0 = row + ((lane >> 4) << 3);
#pragma unroll
  for (int r = 0; r < 8; ++r) dst[(size_t)(r0 + r) * ld + n] = c[r];
}

// window-ordered token g -> (image b, flat index l)
__device__ __forceinline__ void g_to_bl(int g, int& b, int& l) {
  int b_ = g >> 8, n = g & 255;
  b = b_ >> 4;
  int wid = b_ & 15;
  int wy = wid >> 2, wx = wid & 3;
  int ry = n >> 4, rx = n & 15;
  l = (wy * REGION + ry) * 64 + (wx * REGION + rx);
}

// ---------------- prep kernels ----------------
__global__ void prep_wqkv(const float* __restrict__ w, _Float16* __restrict__ wt) {
  int i = blockIdx.x * 256 + threadIdx.x;
  if (i < DIM * QKV_N) {
    int k = i / QKV_N, n = i % QKV_N;
    wt[i] = (_Float16)w[(size_t)n * DIM + k];
  }
}

__global__ void prep_wproj(const float* __restrict__ w, _Float16* __restrict__ wt) {
  int i = blockIdx.x * 256 + threadIdx.x;
  if (i < DIM * DIM) {
    int k = i / DIM, n = i % DIM;
    wt[i] = (_Float16)w[(size_t)n * DIM + k];
  }
}

__global__ void prep_bias(const float* __restrict__ table,
                          const int* __restrict__ idx, float* __restrict__ bias) {
  int i = blockIdx.x * 256 + threadIdx.x;
  if (i < HEADS * NTOK * NTOK) {
    int h = i >> 16, nm = i & 65535;
    bias[i] = table[(size_t)idx[nm] * HEADS + h];
  }
}

// ---------------- K1: QKV GEMM (region-partition fused into addressing) ----------------
__global__ __launch_bounds__(256)
void qkv_gemm(const float* __restrict__ x, const _Float16* __restrict__ wt,
              const float* __restrict__ qkvb, _Float16* __restrict__ Qd,
              _Float16* __restrict__ Kd, _Float16* __restrict__ Vtd) {
  __shared__ __align__(16) _Float16 As[128][32];
  __shared__ __align__(16) _Float16 Bs[128][32];

  int bm = blockIdx.x;
  int bn = blockIdx.y;
  int tid = threadIdx.x, lane = tid & 31, w = tid >> 5;
  int wm = (w & 3) * 32;
  int wn = (w >> 2) * 64;

  v8f acc[2][4];
#pragma unroll
  for (int i = 0; i < 2; ++i)
#pragma unroll
    for (int j = 0; j < 4; ++j) acc[i][j] = zero8();

  int arow = tid >> 1, ahalf = tid & 1;
  int bb, ll;
  g_to_bl(bm * 128 + arow, bb, ll);
  const float* asrc_base = x + ((size_t)bb * L_TOK + ll) * DIM + ahalf * 16;
  int bk = tid >> 3, bnc = (tid & 7) * 16;

  for (int k0 = 0; k0 < DIM; k0 += 32) {
    // stage A (f32 -> f16 in flight, must route through VGPRs)
    {
      const float4* src = (const float4*)(asrc_base + k0);
      _Float16 tmp[16];
#pragma unroll
      for (int i = 0; i < 4; ++i) {
        float4 f = src[i];
        tmp[4 * i + 0] = (_Float16)f.x;
        tmp[4 * i + 1] = (_Float16)f.y;
        tmp[4 * i + 2] = (_Float16)f.z;
        tmp[4 * i + 3] = (_Float16)f.w;
      }
      uint4* d = (uint4*)&As[arow][ahalf * 16];
      d[0] = ((const uint4*)tmp)[0];
      d[1] = ((const uint4*)tmp)[1];
    }
    // stage B transposed: Bs[n][k] = wt[k0+k][bn*128+n]
    {
      const _Float16* src = wt + (size_t)(k0 + bk) * QKV_N + bn * 128 + bnc;
#pragma unroll
      for (int i = 0; i < 16; ++i) Bs[bnc + i][bk] = src[i];
    }
    __syncthreads();

    if (k0 + 32 < DIM) {
      __builtin_prefetch(asrc_base + k0 + 32, 0, 1);
      __builtin_prefetch(wt + (size_t)(k0 + 32 + bk) * QKV_N + bn * 128 + bnc, 0, 1);
    }

    v16h a0 = lda16(&As[0][0], wm, 32, 0, lane);
    v16h a1 = lda16(&As[0][0], wm + 16, 32, 0, lane);
#pragma unroll
    for (int j = 0; j < 4; ++j) {
      v16h bf = ldb16(&Bs[0][0], wn + j * 16, 32, 0, lane);
      acc[0][j] = wmma32(a0, bf, acc[0][j]);
      acc[1][j] = wmma32(a1, bf, acc[1][j]);
    }
    __syncthreads();
  }

#pragma unroll
  for (int mi = 0; mi < 2; ++mi)
#pragma unroll
    for (int j = 0; j < 4; ++j) {
      int c = bn * 128 + wn + j * 16 + (lane & 15);
      float bias = qkvb[c];
      int rbase = bm * 128 + wm + mi * 16 + ((lane >> 4) << 3);
#pragma unroll
      for (int r = 0; r < 8; ++r) {
        int g = rbase + r;
        int b_ = g >> 8, n = g & 255;
        float val = acc[mi][j][r] + bias;
        if (c < 512) {
          int h = c >> 6, d = c & 63;
          Qd[(((size_t)b_ * HEADS + h) * NTOK + n) * HD + d] =
              (_Float16)(val * SCALE);
        } else if (c < 1024) {
          int c2 = c - 512, h = c2 >> 6, d = c2 & 63;
          Kd[(((size_t)b_ * HEADS + h) * NTOK + n) * HD + d] = (_Float16)val;
        } else {
          int c2 = c - 1024, h = c2 >> 6, d = c2 & 63;
          Vtd[(((size_t)b_ * HEADS + h) * HD + d) * NTOK + n] = (_Float16)val;
        }
      }
    }
}

// ---------------- K2: fused attention per (window, head) ----------------
// LDS: S[256][256] f32 (256KB) | Qs[256][64] f16 (32KB) | Ks[256][64] f16 (32KB)
// Q/K/V tiles arrive via TDM; V TDM overlaps conv+softmax, lands in reused Q region.
#define ATTN_LDS (NTOK * NTOK * 4 + 2 * NTOK * HD * 2)

__global__ __launch_bounds__(256)
void region_attn(const _Float16* __restrict__ Qd, const _Float16* __restrict__ Kd,
                 const _Float16* __restrict__ Vtd,
                 const float* __restrict__ bias_pre,
                 const float* __restrict__ pe_w, const float* __restrict__ pe_b,
                 _Float16* __restrict__ Oh) {
  extern __shared__ __align__(16) char smem[];
  float*    S  = (float*)smem;
  _Float16* Qs = (_Float16*)(smem + NTOK * NTOK * 4);
  _Float16* Ks = Qs + NTOK * HD;
  _Float16* Vs = Qs;  // reuse after phase 1

  int bh = blockIdx.x;
  int b_ = bh >> 3, h = bh & 7;
  int tid = threadIdx.x, lane = tid & 31, w = tid >> 5;
  int row0 = w * 32;

  // -------- TDM load of Q, K tiles (wave 0 drives the DMA) --------
  if (w == 0) {
    tdm_copy_f16(Qs, Qd + (size_t)bh * NTOK * HD, NTOK * HD);
    tdm_copy_f16(Ks, Kd + (size_t)bh * NTOK * HD, NTOK * HD);
    tdm_wait0();
  }
  __syncthreads();

  // -------- phase 1: S = Qs * Ks^T --------
  for (int ct = 0; ct < 4; ++ct) {
    int col0 = ct * 64;
    v8f acc[2][4];
#pragma unroll
    for (int i = 0; i < 2; ++i)
#pragma unroll
      for (int j = 0; j < 4; ++j) acc[i][j] = zero8();
#pragma unroll
    for (int ks = 0; ks < 2; ++ks) {
      v16h a0 = lda16(Qs, row0, HD, ks * 32, lane);
      v16h a1 = lda16(Qs, row0 + 16, HD, ks * 32, lane);
#pragma unroll
      for (int j = 0; j < 4; ++j) {
        v16h bf = ldb16(Ks, col0 + j * 16, HD, ks * 32, lane);
        acc[0][j] = wmma32(a0, bf, acc[0][j]);
        acc[1][j] = wmma32(a1, bf, acc[1][j]);
      }
    }
#pragma unroll
    for (int mi = 0; mi < 2; ++mi)
#pragma unroll
      for (int j = 0; j < 4; ++j)
        stacc(S, row0 + mi * 16, NTOK, col0 + j * 16, lane, acc[mi][j]);
  }
  __syncthreads();

  // V TDM issued now: DMA overlaps the whole conv + softmax phase.
  if (w == 0) tdm_copy_f16((void*)Vs, Vtd + (size_t)bh * HD * NTOK, HD * NTOK);

  // -------- phase 2a: depthwise conv over query axis, one column/thread ----
  {
    float wj[CONV_K];
#pragma unroll
    for (int j = 0; j < CONV_K; ++j) wj[j] = pe_w[h * CONV_K + j];
    float pb = pe_b[h];
    int m = tid;
    float win[CONV_K];
#pragma unroll
    for (int i = 0; i < CONV_K; ++i) {
      int nn = i - 7;
      win[i] = (nn >= 0) ? S[(size_t)nn * NTOK + m] : 0.0f;
    }
    for (int n = 0; n < NTOK; ++n) {
      float pe = pb;
#pragma unroll
      for (int j = 0; j < CONV_K; ++j) pe += wj[j] * win[j];
      float outv = win[7] + pe;
#pragma unroll
      for (int j = 0; j < CONV_K - 1; ++j) win[j] = win[j + 1];
      int nn = n + 8;
      win[CONV_K - 1] = (nn < NTOK) ? S[(size_t)nn * NTOK + m] : 0.0f;
      S[(size_t)n * NTOK + m] = outv;
    }
  }
  __syncthreads();

  // -------- phase 2b: + rel-pos bias, softmax over keys (one row per wave) ---
  for (int n = w; n < NTOK; n += 8) {
    const float* bi = bias_pre + ((size_t)h * NTOK + n) * NTOK;
    float vals[8], mx = -3.0e38f;
#pragma unroll
    for (int i = 0; i < 8; ++i) {
      int m = lane + 32 * i;
      float v = S[(size_t)n * NTOK + m] + bi[m];
      vals[i] = v;
      mx = fmaxf(mx, v);
    }
#pragma unroll
    for (int off = 16; off > 0; off >>= 1) mx = fmaxf(mx, __shfl_xor(mx, off, 32));
    float sum = 0.0f;
#pragma unroll
    for (int i = 0; i < 8; ++i) { vals[i] = __expf(vals[i] - mx); sum += vals[i]; }
#pragma unroll
    for (int off = 16; off > 0; off >>= 1) sum += __shfl_xor(sum, off, 32);
    float inv = 1.0f / sum;
#pragma unroll
    for (int i = 0; i < 8; ++i) S[(size_t)n * NTOK + lane + 32 * i] = vals[i] * inv;
  }
  __syncthreads();

  // V TDM must have landed before P@V.
  if (w == 0) tdm_wait0();
  __syncthreads();

  // -------- phase 3: O = P @ V (P f32->f16 fragments on the fly) --------
  {
    v8f acc[2][4];
#pragma unroll
    for (int i = 0; i < 2; ++i)
#pragma unroll
      for (int j = 0; j < 4; ++j) acc[i][j] = zero8();
#pragma unroll
    for (int ks = 0; ks < 8; ++ks) {
      int k0 = ks * 32;
      v16h a0 = lda32(S, row0, NTOK, k0, lane);
      v16h a1 = lda32(S, row0 + 16, NTOK, k0, lane);
#pragma unroll
      for (int j = 0; j < 4; ++j) {
        v16h bf = ldb16(Vs, j * 16, NTOK, k0, lane);
        acc[0][j] = wmma32(a0, bf, acc[0][j]);
        acc[1][j] = wmma32(a1, bf, acc[1][j]);
      }
    }
#pragma unroll
    for (int mi = 0; mi < 2; ++mi)
#pragma unroll
      for (int j = 0; j < 4; ++j) {
        int d = j * 16 + (lane & 15);
        int nb = row0 + mi * 16 + ((lane >> 4) << 3);
#pragma unroll
        for (int r = 0; r < 8; ++r) {
          int n = nb + r;
          Oh[((size_t)b_ * NTOK + n) * DIM + h * HD + d] =
              (_Float16)acc[mi][j][r];
        }
      }
  }
}

// ---------------- K3: proj GEMM + region-reverse scatter ----------------
__global__ __launch_bounds__(256)
void proj_gemm(const _Float16* __restrict__ Oh, const _Float16* __restrict__ Pt,
               const float* __restrict__ pb, float* __restrict__ out) {
  __shared__ __align__(16) _Float16 As[128][32];
  __shared__ __align__(16) _Float16 Bs[128][32];

  int bm = blockIdx.x;
  int bn = blockIdx.y;
  int tid = threadIdx.x, lane = tid & 31, w = tid >> 5;
  int wm = (w & 3) * 32, wn = (w >> 2) * 64;

  v8f acc[2][4];
#pragma unroll
  for (int i = 0; i < 2; ++i)
#pragma unroll
    for (int j = 0; j < 4; ++j) acc[i][j] = zero8();

  int arow = tid >> 1, ahalf = tid & 1;
  const _Float16* asrc_base =
      Oh + (size_t)(bm * 128 + arow) * DIM + ahalf * 16;
  int bk = tid >> 3, bnc = (tid & 7) * 16;

  for (int k0 = 0; k0 < DIM; k0 += 32) {
    // stage A via async global->LDS copy (no VGPR round trip); two 16B chunks
    {
      char* d = (char*)&As[arow][ahalf * 16];
      const char* s = (const char*)(asrc_base + k0);
      async_g2l_b128(d, s);
      async_g2l_b128(d + 16, s + 16);
    }
    // stage B transposed (manual: async can't transpose)
    {
      const _Float16* src = Pt + (size_t)(k0 + bk) * DIM + bn * 128 + bnc;
#pragma unroll
      for (int i = 0; i < 16; ++i) Bs[bnc + i][bk] = src[i];
    }
    async_wait0();
    __syncthreads();

    if (k0 + 32 < DIM) {
      __builtin_prefetch(asrc_base + k0 + 32, 0, 1);
      __builtin_prefetch(Pt + (size_t)(k0 + 32 + bk) * DIM + bn * 128 + bnc, 0, 1);
    }

    v16h a0 = lda16(&As[0][0], wm, 32, 0, lane);
    v16h a1 = lda16(&As[0][0], wm + 16, 32, 0, lane);
#pragma unroll
    for (int j = 0; j < 4; ++j) {
      v16h bf = ldb16(&Bs[0][0], wn + j * 16, 32, 0, lane);
      acc[0][j] = wmma32(a0, bf, acc[0][j]);
      acc[1][j] = wmma32(a1, bf, acc[1][j]);
    }
    __syncthreads();
  }

#pragma unroll
  for (int mi = 0; mi < 2; ++mi)
#pragma unroll
    for (int j = 0; j < 4; ++j) {
      int c = bn * 128 + wn + j * 16 + (lane & 15);
      float bias = pb[c];
      int rbase = bm * 128 + wm + mi * 16 + ((lane >> 4) << 3);
#pragma unroll
      for (int r = 0; r < 8; ++r) {
        int g = rbase + r;
        int b, l;
        g_to_bl(g, b, l);
        out[((size_t)b * L_TOK + l) * DIM + c] = acc[mi][j][r] + bias;
      }
    }
}

// ---------------- host launcher ----------------
extern "C" void kernel_launch(void* const* d_in, const int* in_sizes, int n_in,
                              void* d_out, int out_size, void* d_ws,
                              size_t ws_size, hipStream_t stream) {
  (void)in_sizes; (void)n_in; (void)out_size; (void)ws_size;

  const float* x          = (const float*)d_in[0];
  const float* qkv_w      = (const float*)d_in[1];
  const float* qkv_b      = (const float*)d_in[2];
  const float* proj_w     = (const float*)d_in[3];
  const float* proj_b     = (const float*)d_in[4];
  const float* bias_table = (const float*)d_in[5];
  const float* pe_w       = (const float*)d_in[6];
  const float* pe_b       = (const float*)d_in[7];
  const int*   rel_index  = (const int*)d_in[8];

  char* ws = (char*)d_ws;
  size_t off = 0;
  auto take = [&](size_t bytes) {
    char* p = ws + off;
    off += (bytes + 255) & ~(size_t)255;
    return p;
  };
  _Float16* wt_qkv = (_Float16*)take((size_t)DIM * QKV_N * 2);
  _Float16* wt_prj = (_Float16*)take((size_t)DIM * DIM * 2);
  float*    biasp  = (float*)take((size_t)HEADS * NTOK * NTOK * 4);
  _Float16* Qd     = (_Float16*)take((size_t)NWIN * HEADS * NTOK * HD * 2);
  _Float16* Kd     = (_Float16*)take((size_t)NWIN * HEADS * NTOK * HD * 2);
  _Float16* Vtd    = (_Float16*)take((size_t)NWIN * HEADS * HD * NTOK * 2);
  _Float16* Oh     = (_Float16*)take((size_t)M_TOT * DIM * 2);

  prep_wqkv<<<(DIM * QKV_N + 255) / 256, 256, 0, stream>>>(qkv_w, wt_qkv);
  prep_wproj<<<(DIM * DIM + 255) / 256, 256, 0, stream>>>(proj_w, wt_prj);
  prep_bias<<<(HEADS * NTOK * NTOK + 255) / 256, 256, 0, stream>>>(
      bias_table, rel_index, biasp);

  qkv_gemm<<<dim3(M_TOT / 128, QKV_N / 128), 256, 0, stream>>>(
      x, wt_qkv, qkv_b, Qd, Kd, Vtd);

  (void)hipFuncSetAttribute((const void*)region_attn,
                            hipFuncAttributeMaxDynamicSharedMemorySize,
                            ATTN_LDS);
  region_attn<<<NWIN * HEADS, 256, ATTN_LDS, stream>>>(Qd, Kd, Vtd, biasp,
                                                       pe_w, pe_b, Oh);

  proj_gemm<<<dim3(M_TOT / 128, DIM / 128), 256, 0, stream>>>(
      Oh, wt_prj, proj_b, (float*)d_out);
}
